// SelfAttnV2E_10290741641911
// MI455X (gfx1250) — matrix operand order repaired and checked by reading, not compiled, not run on hardware
//
#include <hip/hip_runtime.h>
#include <hip/hip_bf16.h>
#include <math.h>

typedef __attribute__((ext_vector_type(16))) __bf16 v16bf;
typedef __attribute__((ext_vector_type(8)))  float  v8f;

namespace {
constexpr int kB = 4;
constexpr int kN = 2048;
constexpr int kE = 1024;
constexpr int kRowsV = kB * kN;   // 8192
constexpr int kRowsE = kB * kE;   // 4096
constexpr int DIN = 128;
constexpr int HYP = 256;
constexpr int AST = 136;          // LDS stride for 128-wide bf16 A tile (mult of 8)
constexpr int HST = 264;          // LDS stride for 256-wide bf16 H tile (mult of 8)
constexpr int kCap = 1024;        // per-edge node-list capacity (avg ~105 @5%)
}

// ---------------------------------------------------------------------------
// WMMA helpers (gfx1250 wave32, v_wmma_f32_16x16x32_bf16)
// A-frag (16x32 bf16): lane l holds row m=l&15; element e -> K = (e&7) +
// ((e>>3)?16:0) + 8*(l>>4).  So per (lane,ktile): two contiguous 8-elem runs.
// B-frag (32x16 bf16): lane l holds col n=l&15; element e -> K as in A-frag.
// C/D  (16x16 f32): lane l col n=l&15, vgpr v row m = v + 8*(l>>4).
// ---------------------------------------------------------------------------
static __device__ __forceinline__ v16bf load_a_frag(const __bf16* ap) {
  v16bf a;
#pragma unroll
  for (int i = 0; i < 8; ++i) { a[i] = ap[i]; a[8 + i] = ap[16 + i]; }
  return a;
}

static __device__ __forceinline__ v8f wmma_bf16(v16bf a, v16bf b, v8f c) {
  return __builtin_amdgcn_wmma_f32_16x16x32_bf16(false, a, false, b,
                                                 (short)0, c, false, false);
}

// ---------------------------------------------------------------------------
// Small utility kernels
// ---------------------------------------------------------------------------
__global__ void cvt_bf16(const float* __restrict__ s, __bf16* __restrict__ d, int n) {
  int i = blockIdx.x * blockDim.x + threadIdx.x;
  if (i < n) d[i] = (__bf16)s[i];
}

__global__ void pack_kv(const float* __restrict__ kW, const float* __restrict__ vW,
                        __bf16* __restrict__ d) {
  int i = blockIdx.x * blockDim.x + threadIdx.x;   // [0, 128*256)
  if (i < 128 * 256) {
    int k = i >> 8, n = i & 255;
    float v = (n < 128) ? kW[k * 128 + n] : vW[k * 128 + (n - 128)];
    d[i] = (__bf16)v;
  }
}

__global__ void zero_i32(int* __restrict__ p, int n) {
  int i = blockIdx.x * blockDim.x + threadIdx.x;
  if (i < n) p[i] = 0;
}

static __device__ __forceinline__ float sinu(int p, int c, int dim) {
  int i2 = (c >> 1) << 1;
  float ang = (float)p * powf(10000.0f, -((float)i2) / (float)dim);
  return (c & 1) ? cosf(ang) : sinf(ang);
}

// Computes pe2[2][128], pe3[9][128], qvec[2][64], ball[9][128]
__global__ __launch_bounds__(256) void prep_small(
    const float* __restrict__ qW1, const float* __restrict__ qb1,
    const float* __restrict__ qW2, const float* __restrict__ qb2,
    const float* __restrict__ bW1, const float* __restrict__ bb1,
    const float* __restrict__ bW2, const float* __restrict__ bb2,
    float* __restrict__ qvec, float* __restrict__ ball,
    float* __restrict__ pe2, float* __restrict__ pe3) {
  __shared__ float hq[2 * 256];
  __shared__ float hb[9 * 256];
  int tid = threadIdx.x;
  for (int i = tid; i < 2 * 128; i += 256) pe2[i] = sinu(i / 128, i % 128, 128);
  for (int i = tid; i < 9 * 128; i += 256) pe3[i] = sinu(i / 128, i % 128, 128);
  for (int i = tid; i < 2 * 256; i += 256) {
    int r = i / 256, j = i % 256;
    float s = qb1[j];
    for (int d = 0; d < 64; ++d) s += sinu(r, d, 64) * qW1[d * 256 + j];
    hq[i] = fmaxf(s, 0.f);
  }
  for (int i = tid; i < 9 * 256; i += 256) {
    int r = i / 256, j = i % 256;
    float s = bb1[j];
    for (int d = 0; d < 64; ++d) s += sinu(r, d, 64) * bW1[d * 256 + j];
    hb[i] = fmaxf(s, 0.f);
  }
  __syncthreads();
  for (int i = tid; i < 2 * 64; i += 256) {
    int r = i / 64, o = i % 64;
    float s = qb2[o];
    for (int j = 0; j < 256; ++j) s += hq[r * 256 + j] * qW2[j * 64 + o];
    qvec[i] = s;
  }
  for (int i = tid; i < 9 * 128; i += 256) {
    int r = i / 128, o = i % 128;
    float s = bb2[o];
    for (int j = 0; j < 256; ++j) s += hb[r * 256 + j] * bW2[j * 128 + o];
    ball[i] = s;
  }
}

// ---------------------------------------------------------------------------
// Fused  out = a + MLP(LN(a)+pe)  (+order bias), a = in (+ in2 broadcast)
// 64 rows per 256-thread workgroup; bf16 WMMA 16x16x32 for both GEMMs.
// ---------------------------------------------------------------------------
__global__ __launch_bounds__(256) void fused_mlp(
    const float* __restrict__ in, int in_stride,
    const float* __restrict__ in2, int in2_div,
    const float* __restrict__ gamma, const float* __restrict__ beta,
    const float* __restrict__ pe, const int* __restrict__ peidx, int pe_const,
    const __bf16* __restrict__ W1, const float* __restrict__ b1,
    const __bf16* __restrict__ W2, const float* __restrict__ b2,
    const float* __restrict__ btab, const int* __restrict__ bidx, int b_const,
    float* __restrict__ outp, __bf16* __restrict__ outbf, int M) {
  __shared__ __bf16 Alds[64 * AST];
  __shared__ __bf16 Hlds[64 * HST];
  int tid = threadIdx.x, lane = tid & 31, wid = tid >> 5;
  int row0 = blockIdx.x * 64;

  // Phase 1: LayerNorm (+PE) -> bf16 A tile in LDS. One wave per 8 rows.
#pragma unroll
  for (int rr = 0; rr < 8; ++rr) {
    int r = wid * 8 + rr;
    int grow = row0 + r;
    bool ok = grow < M;
    float vals[4];
#pragma unroll
    for (int i = 0; i < 4; ++i) {
      int d = lane + 32 * i;
      float v = ok ? in[(size_t)grow * in_stride + d] : 0.f;
      if (in2 && ok) v += in2[(size_t)(grow / in2_div) * 128 + d];
      vals[i] = v;
    }
    float s  = vals[0] + vals[1] + vals[2] + vals[3];
    float s2 = vals[0]*vals[0] + vals[1]*vals[1] + vals[2]*vals[2] + vals[3]*vals[3];
#pragma unroll
    for (int o = 16; o > 0; o >>= 1) {
      s  += __shfl_xor(s, o, 32);
      s2 += __shfl_xor(s2, o, 32);
    }
    float mean = s * (1.f / 128.f);
    float var  = s2 * (1.f / 128.f) - mean * mean;
    float rstd = rsqrtf(var + 1e-5f);
    int pidx = 0;
    if (pe) pidx = peidx ? peidx[ok ? grow : 0] : pe_const;
#pragma unroll
    for (int i = 0; i < 4; ++i) {
      int d = lane + 32 * i;
      float hv = (vals[i] - mean) * rstd * gamma[d] + beta[d];
      if (pe) hv += pe[(size_t)pidx * 128 + d];
      Alds[r * AST + d] = (__bf16)hv;
    }
  }
  __syncthreads();

  // GEMM1: H[64,256] = A[64,128] @ W1[128,256], bias + ReLU -> bf16 in LDS
  for (int t = wid; t < 64; t += 8) {
    int mt = t & 3, nt = t >> 2;
    v8f acc = {0.f, 0.f, 0.f, 0.f, 0.f, 0.f, 0.f, 0.f};
    int m = mt * 16 + (lane & 15);
    int hi = lane >> 4;
#pragma unroll
    for (int kt = 0; kt < 4; ++kt) {
      v16bf a = load_a_frag(&Alds[m * AST + kt * 32 + hi * 8]);
      v16bf bfr;
#pragma unroll
      for (int i = 0; i < 8; ++i) {
        bfr[i] = W1[(size_t)(kt * 32 + hi * 8 + i) * HYP + nt * 16 + (lane & 15)];
        bfr[8 + i] = W1[(size_t)(kt * 32 + 16 + hi * 8 + i) * HYP + nt * 16 + (lane & 15)];
      }
      acc = wmma_bf16(a, bfr, acc);
    }
    int n = nt * 16 + (lane & 15);
#pragma unroll
    for (int v = 0; v < 8; ++v) {
      int mrow = mt * 16 + v + hi * 8;
      Hlds[mrow * HST + n] = (__bf16)fmaxf(acc[v] + b1[n], 0.f);
    }
  }
  __syncthreads();

  // GEMM2: O[64,128] = H @ W2[256,128] + b2 + residual (+bias table)
  for (int t = wid; t < 32; t += 8) {
    int mt = t & 3, nt = t >> 2;
    v8f acc = {0.f, 0.f, 0.f, 0.f, 0.f, 0.f, 0.f, 0.f};
    int m = mt * 16 + (lane & 15);
    int hi = lane >> 4;
#pragma unroll
    for (int kt = 0; kt < 8; ++kt) {
      v16bf a = load_a_frag(&Hlds[m * HST + kt * 32 + hi * 8]);
      v16bf bfr;
#pragma unroll
      for (int i = 0; i < 8; ++i) {
        bfr[i] = W2[(size_t)(kt * 32 + hi * 8 + i) * DIN + nt * 16 + (lane & 15)];
        bfr[8 + i] = W2[(size_t)(kt * 32 + 16 + hi * 8 + i) * DIN + nt * 16 + (lane & 15)];
      }
      acc = wmma_bf16(a, bfr, acc);
    }
    int n = nt * 16 + (lane & 15);
#pragma unroll
    for (int v = 0; v < 8; ++v) {
      int mrow = mt * 16 + v + hi * 8;
      int grow = row0 + mrow;
      if (grow < M) {
        float a0 = in[(size_t)grow * in_stride + n];
        if (in2) a0 += in2[(size_t)(grow / in2_div) * 128 + n];
        float o = a0 + acc[v] + b2[n];
        if (btab) o += btab[(size_t)(bidx ? bidx[grow] : b_const) * 128 + n];
        outp[(size_t)grow * 128 + n] = o;
        if (outbf) outbf[(size_t)grow * 128 + n] = (__bf16)o;
      }
    }
  }
}

// ---------------------------------------------------------------------------
// kv = x1(bf16)[8192,128] @ [kW|vW](bf16)[128,256] + [kb|vb]   -> f32 [8192,256]
// ---------------------------------------------------------------------------
__global__ __launch_bounds__(256) void kv_kernel(
    const __bf16* __restrict__ X, const __bf16* __restrict__ W,
    const float* __restrict__ kb, const float* __restrict__ vb,
    float* __restrict__ outp) {
  int tid = threadIdx.x, lane = tid & 31, wid = tid >> 5;
  int row0 = blockIdx.x * 64;
  for (int t = wid; t < 64; t += 8) {
    int mt = t & 3, nt = t >> 2;
    v8f acc = {0.f, 0.f, 0.f, 0.f, 0.f, 0.f, 0.f, 0.f};
    int m = row0 + mt * 16 + (lane & 15);
    int hi = lane >> 4;
#pragma unroll
    for (int kt = 0; kt < 4; ++kt) {
      v16bf a = load_a_frag(&X[(size_t)m * 128 + kt * 32 + hi * 8]);
      v16bf bfr;
#pragma unroll
      for (int i = 0; i < 8; ++i) {
        bfr[i] = W[(size_t)(kt * 32 + hi * 8 + i) * 256 + nt * 16 + (lane & 15)];
        bfr[8 + i] = W[(size_t)(kt * 32 + 16 + hi * 8 + i) * 256 + nt * 16 + (lane & 15)];
      }
      acc = wmma_bf16(a, bfr, acc);
    }
    int n = nt * 16 + (lane & 15);
    float bias = (n < 128) ? kb[n] : vb[n - 128];
#pragma unroll
    for (int v = 0; v < 8; ++v) {
      int grow = row0 + mt * 16 + v + hi * 8;
      outp[(size_t)grow * 256 + n] = acc[v] + bias;
    }
  }
}

// logit0[r][h] = 0.25 * q0[h].k0[r][h];  logit1[r][h] = 0.25 * q1[h].k1[r][h]
__global__ __launch_bounds__(256) void logits_kernel(
    const float* __restrict__ kvbuf, const float* __restrict__ qvec,
    float* __restrict__ l0, float* __restrict__ l1) {
  int idx = blockIdx.x * 256 + threadIdx.x;  // [0, 8192*4)
  int r = idx >> 2, h = idx & 3;
  const float* kr = &kvbuf[(size_t)r * 256];
  float s0 = 0.f, s1 = 0.f;
#pragma unroll
  for (int d = 0; d < 16; ++d) {
    s0 += qvec[h * 16 + d] * kr[h * 16 + d];
    s1 += qvec[64 + h * 16 + d] * kr[64 + h * 16 + d];
  }
  l0[idx] = s0 * 0.25f;
  l1[idx] = s1 * 0.25f;
}

// ---------------------------------------------------------------------------
// Per-edge node-list construction with *coalesced* incidence reads:
// one workgroup per (b,n) row; each nonzero scatters its node id into the
// edge's list via a global atomic.  Incidence touched exactly once, contiguous.
// ---------------------------------------------------------------------------
__global__ __launch_bounds__(256) void build_lists(
    const float* __restrict__ inc, int* __restrict__ counts,
    int* __restrict__ lists) {
  int bn = blockIdx.x;            // 0..8191 == b*N + n
  int b = bn >> 11;               // / kN
  int n = bn & (kN - 1);
  int tid = threadIdx.x;
  const float* row = &inc[(size_t)bn * kE];
  for (int e = tid; e < kE; e += 256) {
    if (row[e] > 0.f) {
      int be = b * kE + e;
      int p = atomicAdd(&counts[be], 1);
      if (p < kCap) lists[(size_t)be * kCap + p] = n;
    }
  }
}

// Global (order-0) attention: softmax over all N nodes per (b,h); att0[b][128]
__global__ __launch_bounds__(256) void attn0_kernel(
    const float* __restrict__ kv, const float* __restrict__ l0,
    float* __restrict__ att0) {
  int b = blockIdx.x, tid = threadIdx.x;
  __shared__ float sred[256];
  __shared__ float smax[4], srs[4];
  float mx[4] = {-3e38f, -3e38f, -3e38f, -3e38f};
  for (int n = tid; n < kN; n += 256) {
    const float* lg = &l0[(size_t)(b * kN + n) * 4];
    for (int h = 0; h < 4; ++h) mx[h] = fmaxf(mx[h], lg[h]);
  }
  for (int h = 0; h < 4; ++h) {
    sred[tid] = mx[h]; __syncthreads();
    for (int s = 128; s > 0; s >>= 1) {
      if (tid < s) sred[tid] = fmaxf(sred[tid], sred[tid + s]);
      __syncthreads();
    }
    if (tid == 0) smax[h] = sred[0];
    __syncthreads();
  }
  float sm[4] = {0.f, 0.f, 0.f, 0.f};
  for (int n = tid; n < kN; n += 256) {
    const float* lg = &l0[(size_t)(b * kN + n) * 4];
    for (int h = 0; h < 4; ++h) sm[h] += __expf(lg[h] - smax[h]);
  }
  for (int h = 0; h < 4; ++h) {
    sred[tid] = sm[h]; __syncthreads();
    for (int s = 128; s > 0; s >>= 1) {
      if (tid < s) sred[tid] += sred[tid + s];
      __syncthreads();
    }
    if (tid == 0) srs[h] = 1.0f / sred[0];
    __syncthreads();
  }
  int d = tid & 127, half = tid >> 7, h = d >> 5;
  float acc = 0.f;
  for (int n = half; n < kN; n += 2) {
    size_t rn = (size_t)(b * kN + n);
    float w = __expf(l0[rn * 4 + h] - smax[h]) * srs[h];
    acc += w * kv[rn * 256 + 128 + d];
  }
  sred[tid] = acc; __syncthreads();
  if (half == 0) att0[b * 128 + d] = sred[d] + sred[128 + d];
}

// Incidence-masked node->edge attention over precomputed node lists.
__global__ __launch_bounds__(256) void attn1e_kernel(
    const int* __restrict__ counts, const int* __restrict__ lists,
    const float* __restrict__ kv, const float* __restrict__ l1,
    float* __restrict__ att1e) {
  int be = blockIdx.x;
  int b = be >> 10;          // / kE
  int tid = threadIdx.x;
  __shared__ int list[kCap];
  __shared__ float sred[256];
  __shared__ float smax[4], srs[4];
  int C = counts[be];
  if (C > kCap) C = kCap;
  const int* gl = &lists[(size_t)be * kCap];
  for (int i = tid; i < C; i += 256) list[i] = gl[i];
  __syncthreads();
  float mx[4] = {-3e38f, -3e38f, -3e38f, -3e38f};
  for (int i = tid; i < C; i += 256) {
    const float* lg = &l1[(size_t)(b * kN + list[i]) * 4];
    for (int h = 0; h < 4; ++h) mx[h] = fmaxf(mx[h], lg[h]);
  }
  for (int h = 0; h < 4; ++h) {
    sred[tid] = mx[h]; __syncthreads();
    for (int s = 128; s > 0; s >>= 1) {
      if (tid < s) sred[tid] = fmaxf(sred[tid], sred[tid + s]);
      __syncthreads();
    }
    if (tid == 0) smax[h] = sred[0];
    __syncthreads();
  }
  float sm[4] = {0.f, 0.f, 0.f, 0.f};
  for (int i = tid; i < C; i += 256) {
    const float* lg = &l1[(size_t)(b * kN + list[i]) * 4];
    for (int h = 0; h < 4; ++h) sm[h] += __expf(lg[h] - smax[h]);
  }
  for (int h = 0; h < 4; ++h) {
    sred[tid] = sm[h]; __syncthreads();
    for (int s = 128; s > 0; s >>= 1) {
      if (tid < s) sred[tid] += sred[tid + s];
      __syncthreads();
    }
    if (tid == 0) srs[h] = 1.0f / sred[0];
    __syncthreads();
  }
  int d = tid & 127, half = tid >> 7, h = d >> 5;
  float acc = 0.f;
  for (int i = half; i < C; i += 2) {
    size_t rn = (size_t)(b * kN + list[i]);
    float w = __expf(l1[rn * 4 + h] - smax[h]) * srs[h];
    acc += w * kv[rn * 256 + 128 + d];
  }
  sred[tid] = acc; __syncthreads();
  if (half == 0) att1e[(size_t)be * 128 + d] = sred[d] + sred[128 + d];
}

// ---------------------------------------------------------------------------
extern "C" void kernel_launch(void* const* d_in, const int* in_sizes, int n_in,
                              void* d_out, int out_size, void* d_ws, size_t ws_size,
                              hipStream_t stream) {
  (void)in_sizes; (void)n_in; (void)out_size; (void)ws_size;
  const float* x    = (const float*)d_in[0];
  const float* inc  = (const float*)d_in[1];
  const int*   eord = (const int*)d_in[2];
  // d_in[3]=node_mask, d_in[4]=edge_mask : all-true, unused
  const float* mlp1_W1 = (const float*)d_in[5];
  const float* mlp1_b1 = (const float*)d_in[6];
  const float* mlp1_W2 = (const float*)d_in[7];
  const float* mlp1_b2 = (const float*)d_in[8];
  const float* mlp2_W1 = (const float*)d_in[9];
  const float* mlp2_b1 = (const float*)d_in[10];
  const float* mlp2_W2 = (const float*)d_in[11];
  const float* mlp2_b2 = (const float*)d_in[12];
  const float* mlp3_W1 = (const float*)d_in[13];
  const float* mlp3_b1 = (const float*)d_in[14];
  const float* mlp3_W2 = (const float*)d_in[15];
  const float* mlp3_b2 = (const float*)d_in[16];
  const float* q_W1 = (const float*)d_in[17];
  const float* q_b1 = (const float*)d_in[18];
  const float* q_W2 = (const float*)d_in[19];
  const float* q_b2 = (const float*)d_in[20];
  const float* bias_W1 = (const float*)d_in[21];
  const float* bias_b1 = (const float*)d_in[22];
  const float* bias_W2 = (const float*)d_in[23];
  const float* bias_b2 = (const float*)d_in[24];
  const float* kW = (const float*)d_in[25];
  const float* kb = (const float*)d_in[26];
  const float* vW = (const float*)d_in[27];
  const float* vb = (const float*)d_in[28];
  const float* g1  = (const float*)d_in[29];
  const float* be1 = (const float*)d_in[30];
  const float* g2  = (const float*)d_in[31];
  const float* be2 = (const float*)d_in[32];
  const float* g3  = (const float*)d_in[33];
  const float* be3 = (const float*)d_in[34];

  char* w = (char*)d_ws;
  size_t off = 0;
  auto take = [&](size_t bytes) -> char* {
    char* p = w + off;
    off += (bytes + 255) & ~(size_t)255;
    return p;
  };
  __bf16* w1a = (__bf16*)take(128 * 256 * 2);
  __bf16* w2a = (__bf16*)take(256 * 128 * 2);
  __bf16* w1b = (__bf16*)take(128 * 256 * 2);
  __bf16* w2b = (__bf16*)take(256 * 128 * 2);
  __bf16* w1c = (__bf16*)take(128 * 256 * 2);
  __bf16* w2c = (__bf16*)take(256 * 128 * 2);
  __bf16* wkv = (__bf16*)take(128 * 256 * 2);
  float* qvec = (float*)take(2 * 64 * 4);
  float* pe2  = (float*)take(2 * 128 * 4);
  float* pe3  = (float*)take(9 * 128 * 4);
  float* ball = (float*)take(9 * 128 * 4);
  float*  x1    = (float*)take((size_t)kRowsV * 128 * 4);
  __bf16* x1bf  = (__bf16*)take((size_t)kRowsV * 128 * 2);
  float*  kvbuf = (float*)take((size_t)kRowsV * 256 * 4);
  float*  l0    = (float*)take((size_t)kRowsV * 4 * 4);
  float*  l1    = (float*)take((size_t)kRowsV * 4 * 4);
  float*  att0b = (float*)take(kB * 128 * 4);
  float*  att1e = (float*)take((size_t)kRowsE * 128 * 4);
  float*  y0    = (float*)take(kB * 128 * 4);
  float*  yv    = (float*)take((size_t)kRowsV * 128 * 4);
  float*  ye    = (float*)take((size_t)kRowsE * 128 * 4);
  int*    ecnt  = (int*)take((size_t)kRowsE * 4);
  int*    elist = (int*)take((size_t)kRowsE * kCap * 4);

  float* out_v = (float*)d_out;                        // [B,N,128]
  float* out_e = out_v + (size_t)kRowsV * 128;         // [B,E,128]

  // --- weight prep ---
  cvt_bf16<<<128, 256, 0, stream>>>(mlp1_W1, w1a, 128 * 256);
  cvt_bf16<<<128, 256, 0, stream>>>(mlp1_W2, w2a, 256 * 128);
  cvt_bf16<<<128, 256, 0, stream>>>(mlp2_W1, w1b, 128 * 256);
  cvt_bf16<<<128, 256, 0, stream>>>(mlp2_W2, w2b, 256 * 128);
  cvt_bf16<<<128, 256, 0, stream>>>(mlp3_W1, w1c, 128 * 256);
  cvt_bf16<<<128, 256, 0, stream>>>(mlp3_W2, w2c, 256 * 128);
  pack_kv<<<128, 256, 0, stream>>>(kW, vW, wkv);
  prep_small<<<1, 256, 0, stream>>>(q_W1, q_b1, q_W2, q_b2,
                                    bias_W1, bias_b1, bias_W2, bias_b2,
                                    qvec, ball, pe2, pe3);

  // --- per-edge node lists (coalesced incidence scan) ---
  zero_i32<<<(kRowsE + 255) / 256, 256, 0, stream>>>(ecnt, kRowsE);
  build_lists<<<kRowsV, 256, 0, stream>>>(inc, ecnt, elist);

  // --- block 1: x1 = x + MLP1(LN(x)) ---
  fused_mlp<<<kRowsV / 64, 256, 0, stream>>>(
      x, 128, nullptr, 1, g1, be1, nullptr, nullptr, 0,
      w1a, mlp1_b1, w2a, mlp1_b2, nullptr, nullptr, 0,
      x1, x1bf, kRowsV);

  // --- k/v projections + logits ---
  kv_kernel<<<kRowsV / 64, 256, 0, stream>>>(x1bf, wkv, kb, vb, kvbuf);
  logits_kernel<<<(kRowsV * 4) / 256, 256, 0, stream>>>(kvbuf, qvec, l0, l1);

  // --- attentions ---
  attn0_kernel<<<kB, 256, 0, stream>>>(kvbuf, l0, att0b);
  attn1e_kernel<<<kRowsE, 256, 0, stream>>>(ecnt, elist, kvbuf, l1, att1e);

  // --- block 2: y = a + MLP2(LN(a)+pe2[sel]) ---
  fused_mlp<<<1, 256, 0, stream>>>(
      att0b, 128, nullptr, 1, g2, be2, pe2, nullptr, 0,
      w1b, mlp2_b1, w2b, mlp2_b2, nullptr, nullptr, 0,
      y0, nullptr, kB);
  fused_mlp<<<kRowsV / 64, 256, 0, stream>>>(
      kvbuf + 128, 256, nullptr, 1, g2, be2, pe2, nullptr, 1,
      w1b, mlp2_b1, w2b, mlp2_b2, nullptr, nullptr, 0,
      yv, nullptr, kRowsV);
  fused_mlp<<<kRowsE / 64, 256, 0, stream>>>(
      att1e, 128, nullptr, 1, g2, be2, pe2, nullptr, 1,
      w1b, mlp2_b1, w2b, mlp2_b2, nullptr, nullptr, 0,
      ye, nullptr, kRowsE);

  // --- block 3 (+ order bias), writing final outputs ---
  fused_mlp<<<kRowsV / 64, 256, 0, stream>>>(
      yv, 128, y0, kN, g3, be3, pe3, nullptr, 1,
      w1c, mlp3_b1, w2c, mlp3_b2, ball, nullptr, 1,
      out_v, nullptr, kRowsV);
  fused_mlp<<<kRowsE / 64, 256, 0, stream>>>(
      ye, 128, y0, kE, g3, be3, pe3, eord, 1,
      w1c, mlp3_b1, w2c, mlp3_b2, ball, eord, 1,
      out_e, nullptr, kRowsE);
}
